// Bridge_BlockV1_74534862455395
// MI455X (gfx1250) — matrix operand order repaired and checked by reading, not compiled, not run on hardware
//
#include <hip/hip_runtime.h>
#include <math.h>

typedef __attribute__((ext_vector_type(16))) __bf16 v16bf;
typedef __attribute__((ext_vector_type(8)))  __bf16 v8bf;
typedef __attribute__((ext_vector_type(4)))  __bf16 v4bf;
typedef __attribute__((ext_vector_type(8)))  float  v8f;
typedef __attribute__((ext_vector_type(4)))  float  v4f;

#define EPSV      1e-6f
#define K_DIM     4096
#define N_DIM     4096
#define M_HALF    8192
#define M_TOT     16384
#define BM        128
#define BN        128
#define BK        32
#define LDK       40        // padded LDS row (bf16 elems): 80B stride, 16B aligned, 20-bank step
#define TILE      (BM * LDK)        // elems per limb tile (5120)
#define SBUF      (4 * TILE)        // elems per double-buffer slot (Ah,Al,Wh,Wl)
#define SMEM_BYTES (2 * SBUF * 2)   // 81920 bytes dynamic LDS
#define LD2       264       // kernel2 LDS row: 528B stride, 16B aligned
#define CW_ELEMS  65536     // 256*256

__device__ __forceinline__ void split_bf16(float x, __bf16& h, __bf16& l) {
  h = (__bf16)x;
  l = (__bf16)(x - (float)h);
}

__device__ __forceinline__ v16bf cat8(v8bf a, v8bf b) {
  return __builtin_shufflevector(a, b, 0,1,2,3,4,5,6,7,8,9,10,11,12,13,14,15);
}

__device__ __forceinline__ v8f wmma_bf16(v16bf a, v16bf b, v8f c) {
  return __builtin_amdgcn_wmma_f32_16x16x32_bf16(false, a, false, b, (short)0, c, false, false);
}

// ---------------------------------------------------------------------------
// Kernel 0: split c_W (fp32 [256,256], row o, col k) into bf16 hi/lo limbs.
// ---------------------------------------------------------------------------
__global__ void prep_cw_kernel(const float* __restrict__ cW,
                               __bf16* __restrict__ cwh, __bf16* __restrict__ cwl) {
  int i = blockIdx.x * blockDim.x + threadIdx.x;
  if (i < CW_ELEMS) {
    float v = cW[i];
    __bf16 h, l;
    split_bf16(v, h, l);
    cwh[i] = h;
    cwl[i] = l;
  }
}

// ---------------------------------------------------------------------------
// Kernel 1: C[m,n] = sum_k A[m,k]*W[n,k] + r_b[n],  A = [f_r; f_i] (M=16384)
// bf16x2-limb split, 3 WMMAs per 16x16x32 tile step.
// Software-pipelined: next tile's global loads issued before the WMMA block,
// converted + stored into the other LDS buffer after it; 1 barrier / K-step.
// ---------------------------------------------------------------------------
__global__ __launch_bounds__(256)
void gemm_kernel(const float* __restrict__ f_r, const float* __restrict__ f_i,
                 const float* __restrict__ r_W, const float* __restrict__ r_b,
                 float* __restrict__ out) {
  extern __shared__ __bf16 smem[];   // 2 buffers x (Ah|Al|Wh|Wl), 81920 B

  const int tid    = threadIdx.x;
  const int wave   = tid >> 5;
  const int lane   = tid & 31;
  const int l15    = lane & 15;
  const int half8  = (lane >> 4) * 8;    // A-fragment K sub-offset / C-row offset
  const int half16 = (lane >> 4) * 16;   // unused; B fragment uses half8
  const int wm     = wave >> 2;          // 0..1 : 64-row strip
  const int wn     = wave & 3;           // 0..3 : 32-col strip
  const int bn0    = blockIdx.x * BN;
  const int bm0    = blockIdx.y * BM;

  const int srow = tid >> 3;             // 0..31 staging row
  const int scol = (tid & 7) * 4;        // 0,4,...,28 staging col (float4)

  // staging source row pointers (row index is loop-invariant)
  const float* asrc[4];
  const float* wsrc[4];
#pragma unroll
  for (int p = 0; p < 4; ++p) {
    const int r   = srow + p * 32;
    const int m_g = bm0 + r;
    asrc[p] = (m_g < M_HALF) ? (f_r + (size_t)m_g * K_DIM)
                             : (f_i + (size_t)(m_g - M_HALF) * K_DIM);
    wsrc[p] = r_W + (size_t)(bn0 + r) * K_DIM;
  }

  v8f acc[4][2];
#pragma unroll
  for (int mt = 0; mt < 4; ++mt)
#pragma unroll
    for (int nt = 0; nt < 2; ++nt)
      acc[mt][nt] = (v8f){0.f,0.f,0.f,0.f,0.f,0.f,0.f,0.f};

  v4f areg[4], wreg[4];

  // ---- prologue: stage tile 0 into buffer 0 ----
#pragma unroll
  for (int p = 0; p < 4; ++p) {
    areg[p] = *(const v4f*)(asrc[p] + scol);
    wreg[p] = *(const v4f*)(wsrc[p] + scol);
  }
#pragma unroll
  for (int p = 0; p < 4; ++p) {
    const int r = srow + p * 32;
    v4bf ah, al, wh, wl;
#pragma unroll
    for (int e = 0; e < 4; ++e) {
      __bf16 h, l;
      split_bf16(areg[p][e], h, l); ah[e] = h; al[e] = l;
      split_bf16(wreg[p][e], h, l); wh[e] = h; wl[e] = l;
    }
    *(v4bf*)&smem[0 * SBUF + 0 * TILE + r * LDK + scol] = ah;
    *(v4bf*)&smem[0 * SBUF + 1 * TILE + r * LDK + scol] = al;
    *(v4bf*)&smem[0 * SBUF + 2 * TILE + r * LDK + scol] = wh;
    *(v4bf*)&smem[0 * SBUF + 3 * TILE + r * LDK + scol] = wl;
  }
  __syncthreads();

  const int NK = K_DIM / BK;   // 128
  for (int kt = 0; kt < NK; ++kt) {
    const int cur  = kt & 1;
    const bool more = (kt + 1) < NK;

    // ---- issue next tile's global loads early (latency hidden by WMMAs) ----
    if (more) {
      const int k0 = (kt + 1) * BK + scol;
#pragma unroll
      for (int p = 0; p < 4; ++p) {
        areg[p] = *(const v4f*)(asrc[p] + k0);
        wreg[p] = *(const v4f*)(wsrc[p] + k0);
      }
    }

    // ---- fragments from current buffer ----
    const __bf16* Ah = smem + cur * SBUF;
    const __bf16* Al = Ah + TILE;
    const __bf16* Wh = Ah + 2 * TILE;
    const __bf16* Wl = Ah + 3 * TILE;

    // A 16x32 bf16 layout: lane-low elems 0..7 -> K 0..7, 8..15 -> K 16..23;
    //                      lane-high        -> K 8..15 / 24..31.
    v16bf afh[4], afl[4];
#pragma unroll
    for (int mt = 0; mt < 4; ++mt) {
      const int row = (wm * 64 + mt * 16 + l15) * LDK;
      afh[mt] = cat8(*(const v8bf*)&Ah[row + half8], *(const v8bf*)&Ah[row + 16 + half8]);
      afl[mt] = cat8(*(const v8bf*)&Al[row + half8], *(const v8bf*)&Al[row + 16 + half8]);
    }
    // B 32x16 bf16 layout: same K order as A (lane-low K 0..7/16..23, lane-high K 8..15/24..31).
    v16bf bfh[2], bfl[2];
#pragma unroll
    for (int nt = 0; nt < 2; ++nt) {
      const int col = (wn * 32 + nt * 16 + l15) * LDK;
      bfh[nt] = cat8(*(const v8bf*)&Wh[col + half8], *(const v8bf*)&Wh[col + 16 + half8]);
      bfl[nt] = cat8(*(const v8bf*)&Wl[col + half8], *(const v8bf*)&Wl[col + 16 + half8]);
    }

    // ---- 3-limb WMMA: hi*hi + hi*lo + lo*hi ----
#pragma unroll
    for (int mt = 0; mt < 4; ++mt)
#pragma unroll
      for (int nt = 0; nt < 2; ++nt) {
        acc[mt][nt] = wmma_bf16(afh[mt], bfh[nt], acc[mt][nt]);
        acc[mt][nt] = wmma_bf16(afh[mt], bfl[nt], acc[mt][nt]);
        acc[mt][nt] = wmma_bf16(afl[mt], bfh[nt], acc[mt][nt]);
      }

    // ---- convert + store next tile into the other buffer ----
    if (more) {
      __bf16* nAh = smem + (1 - cur) * SBUF;
#pragma unroll
      for (int p = 0; p < 4; ++p) {
        const int r = srow + p * 32;
        v4bf ah, al, wh, wl;
#pragma unroll
        for (int e = 0; e < 4; ++e) {
          __bf16 h, l;
          split_bf16(areg[p][e], h, l); ah[e] = h; al[e] = l;
          split_bf16(wreg[p][e], h, l); wh[e] = h; wl[e] = l;
        }
        *(v4bf*)&nAh[0 * TILE + r * LDK + scol] = ah;
        *(v4bf*)&nAh[1 * TILE + r * LDK + scol] = al;
        *(v4bf*)&nAh[2 * TILE + r * LDK + scol] = wh;
        *(v4bf*)&nAh[3 * TILE + r * LDK + scol] = wl;
      }
    }
    __syncthreads();
  }

  // ---- epilogue: +bias, store. C/D layout: VGPR v -> row v + half8, col = l15 ----
#pragma unroll
  for (int nt = 0; nt < 2; ++nt) {
    const int n_g = bn0 + wn * 32 + nt * 16 + l15;
    const float rb = r_b[n_g];
#pragma unroll
    for (int mt = 0; mt < 4; ++mt) {
      const int m_base = bm0 + wm * 64 + mt * 16 + half8;
#pragma unroll
      for (int v = 0; v < 8; ++v) {
        out[(size_t)(m_base + v) * N_DIM + n_g] = acc[mt][nt][v] + rb;
      }
    }
  }
}

// ---------------------------------------------------------------------------
// Kernel 2: phase branch. One block per batch row b.
//   l = 0.5*log(fr^2+fi^2+2eps)+w_lam ; t = atan2(fi+eps, fr+eps)+w_tha
//   Y = (transposed) @ c_W.T + c_b + bias ; out += exp(Yl)*{cos,sin}(Yt)
// ---------------------------------------------------------------------------
__global__ __launch_bounds__(256)
void phase_kernel(const float* __restrict__ f_r, const float* __restrict__ f_i,
                  const float* __restrict__ c_b,
                  const float* __restrict__ w_lam, const float* __restrict__ w_tha,
                  const float* __restrict__ b_lam, const float* __restrict__ b_tha,
                  const __bf16* __restrict__ cwh, const __bf16* __restrict__ cwl,
                  float* __restrict__ out) {
  __shared__ __bf16 Lh[16 * LD2];
  __shared__ __bf16 Ll[16 * LD2];
  __shared__ __bf16 Th[16 * LD2];
  __shared__ __bf16 Tl[16 * LD2];

  const int b   = blockIdx.x;
  const int tid = threadIdx.x;

  // ---- step 1: elementwise, store transposed (A = lT is 16 x 256) ----
#pragma unroll
  for (int p = 0; p < 16; ++p) {
    const int idx = tid + p * 256;           // = k*16 + j
    const float fr = f_r[(size_t)b * 4096 + idx];
    const float fi = f_i[(size_t)b * 4096 + idx];
    const float lv = 0.5f * __logf(fr * fr + fi * fi + 2.0f * EPSV) + w_lam[idx];
    const float tv = atan2f(fi + EPSV, fr + EPSV) + w_tha[idx];
    const int j = idx & 15;
    const int k = idx >> 4;
    __bf16 h, l;
    split_bf16(lv, h, l); Lh[j * LD2 + k] = h; Ll[j * LD2 + k] = l;
    split_bf16(tv, h, l); Th[j * LD2 + k] = h; Tl[j * LD2 + k] = l;
  }
  __syncthreads();

  const int wave   = tid >> 5;
  const int lane   = tid & 31;
  const int l15    = lane & 15;
  const int half8  = (lane >> 4) * 8;
  const int half16 = (lane >> 4) * 16;

  v8f accl[2], acct[2];
#pragma unroll
  for (int nt = 0; nt < 2; ++nt) {
    accl[nt] = (v8f){0.f,0.f,0.f,0.f,0.f,0.f,0.f,0.f};
    acct[nt] = (v8f){0.f,0.f,0.f,0.f,0.f,0.f,0.f,0.f};
  }

  for (int ks = 0; ks < 256; ks += 32) {
    const int rA = l15 * LD2 + ks;
    v16bf alh = cat8(*(const v8bf*)&Lh[rA + half8], *(const v8bf*)&Lh[rA + 16 + half8]);
    v16bf all = cat8(*(const v8bf*)&Ll[rA + half8], *(const v8bf*)&Ll[rA + 16 + half8]);
    v16bf ath = cat8(*(const v8bf*)&Th[rA + half8], *(const v8bf*)&Th[rA + 16 + half8]);
    v16bf atl = cat8(*(const v8bf*)&Tl[rA + half8], *(const v8bf*)&Tl[rA + 16 + half8]);

#pragma unroll
    for (int nt = 0; nt < 2; ++nt) {
      const int o = (wave * 2 + nt) * 16 + l15;        // output column (row of c_W)
      const __bf16* ph = cwh + (size_t)o * 256 + ks + half8;
      const __bf16* pl = cwl + (size_t)o * 256 + ks + half8;
      v16bf bh = cat8(*(const v8bf*)ph, *(const v8bf*)(ph + 16));
      v16bf bl = cat8(*(const v8bf*)pl, *(const v8bf*)(pl + 16));

      accl[nt] = wmma_bf16(alh, bh, accl[nt]);
      accl[nt] = wmma_bf16(alh, bl, accl[nt]);
      accl[nt] = wmma_bf16(all, bh, accl[nt]);

      acct[nt] = wmma_bf16(ath, bh, acct[nt]);
      acct[nt] = wmma_bf16(ath, bl, acct[nt]);
      acct[nt] = wmma_bf16(atl, bh, acct[nt]);
    }
  }

  // ---- epilogue: +c_b+bias, exp/sincos, transpose back, RMW into d_out ----
  const size_t NT = (size_t)M_HALF * N_DIM;
#pragma unroll
  for (int nt = 0; nt < 2; ++nt) {
    const int o = (wave * 2 + nt) * 16 + l15;
    const float cb = c_b[o];
#pragma unroll
    for (int v = 0; v < 8; ++v) {
      const int j = v + half8;
      const float yl = accl[nt][v] + cb + b_lam[j * 256 + o];
      const float yt = acct[nt][v] + cb + b_tha[j * 256 + o];
      const float lm = __expf(yl);
      float s, c;
      __sincosf(yt, &s, &c);
      const size_t addr = (size_t)b * 4096 + (size_t)o * 16 + j;
      out[addr]      += lm * c;
      out[NT + addr] += lm * s;
    }
  }
}

// ---------------------------------------------------------------------------
extern "C" void kernel_launch(void* const* d_in, const int* in_sizes, int n_in,
                              void* d_out, int out_size, void* d_ws, size_t ws_size,
                              hipStream_t stream) {
  const float* f_r   = (const float*)d_in[0];
  const float* f_i   = (const float*)d_in[1];
  const float* r_W   = (const float*)d_in[2];
  const float* r_b   = (const float*)d_in[3];
  const float* c_W   = (const float*)d_in[4];
  const float* c_b   = (const float*)d_in[5];
  const float* w_lam = (const float*)d_in[6];
  const float* w_tha = (const float*)d_in[7];
  const float* b_lam = (const float*)d_in[8];
  const float* b_tha = (const float*)d_in[9];
  float* out = (float*)d_out;

  __bf16* cwh = (__bf16*)d_ws;          // 256KB of scratch: c_W hi/lo limbs
  __bf16* cwl = cwh + CW_ELEMS;

  prep_cw_kernel<<<CW_ELEMS / 256, 256, 0, stream>>>(c_W, cwh, cwl);
  gemm_kernel<<<dim3(N_DIM / BN, M_TOT / BM), 256, SMEM_BYTES, stream>>>(f_r, f_i, r_W, r_b, out);
  phase_kernel<<<M_HALF, 256, 0, stream>>>(f_r, f_i, c_b, w_lam, w_tha,
                                           b_lam, b_tha, cwh, cwl, out);
}
